// AttentionCritic_31610959299057
// MI455X (gfx1250) — matrix operand order repaired and checked by reading, not compiled, not run on hardware
//
#include <hip/hip_runtime.h>

// ---------------- problem constants ----------------
constexpr int NA   = 8;       // agents
constexpr int NB   = 32768;   // batch
constexpr int SD   = 96;      // SDIM
constexpr int AD   = 32;      // ADIM
constexpr int LD   = 136;     // padded LDS row stride (halves)
constexpr int CLD  = 264;     // padded stride for K=256 tiles

typedef _Float16 h16;
typedef __attribute__((ext_vector_type(16))) _Float16 v16h;
typedef __attribute__((ext_vector_type(8)))  _Float16 v8h;
typedef __attribute__((ext_vector_type(8)))  float    v8f;
typedef __attribute__((ext_vector_type(4)))  int      i4;

__device__ __forceinline__ v8f wmma_f16(v16h a, v16h b, v8f c) {
  return __builtin_amdgcn_wmma_f32_16x16x32_f16(false, a, false, b, (short)0, c,
                                                false, false);
}

// A fragment (16x32 f16) from row-major LDS tile X[m][k] (row stride ld)
__device__ __forceinline__ v16h frag_a(const h16* X, int ld, int m0, int k0, int lane) {
  const h16* p = X + (m0 + (lane & 15)) * ld + k0 + ((lane >> 4) << 4);
  v16h a;
#pragma unroll
  for (int i = 0; i < 16; ++i) a[i] = p[i];
  return a;
}
// B fragment (32x16 f16) from TRANSPOSED weights Wt[n][k] (row stride ld)
__device__ __forceinline__ v16h frag_b(const h16* Wt, int ld, int n0, int k0, int lane) {
  const h16* p = Wt + (n0 + (lane & 15)) * ld + k0 + ((lane >> 4) << 4);
  v16h b;
#pragma unroll
  for (int i = 0; i < 16; ++i) b[i] = p[i];
  return b;
}

__device__ __forceinline__ float lrelu(float x) { return x >= 0.f ? x : 0.01f * x; }

// CDNA5 async global->LDS 16B copy (ASYNCcnt-tracked), with safe fallback.
// Signature (from hipcc diagnostic): (v4i AS1* src, v4i AS3* lds_dst, Ii off, Ii cpol)
__device__ __forceinline__ void copy16_g2l(h16* l, const h16* g) {
#if defined(__gfx1250__) && __has_builtin(__builtin_amdgcn_global_load_async_to_lds_b128)
  void* gv = (void*)g;  // strip const
  __builtin_amdgcn_global_load_async_to_lds_b128(
      (__attribute__((address_space(1))) i4*)gv,
      (__attribute__((address_space(3))) i4*)l, 0, 0);
#else
  *(float4*)l = *(const float4*)g;
#endif
}
__device__ __forceinline__ void wait_async0() {
#if defined(__gfx1250__) && __has_builtin(__builtin_amdgcn_s_wait_asynccnt)
  __builtin_amdgcn_s_wait_asynccnt(0);
#endif
}

// M=128 x N=128 x K GEMM, 8 waves (wave w owns m-block w), optional bias/act,
// writes f16 to LDS (stride LD) and/or global (row stride 128).
// B fragments for a k-step are batch-loaded so ds_load_b128s issue ahead of the
// WMMA chain (overlap LDS latency with matrix ops).
__device__ __forceinline__ void do_gemm(const h16* __restrict__ X, int ldx,
                                        const h16* __restrict__ W, int ldw, int K,
                                        const float* __restrict__ bias, bool act,
                                        h16* outL, h16* __restrict__ outG, long grow,
                                        int wv, int lane) {
  int m0 = wv * 16;
  v8f acc[8];
#pragma unroll
  for (int t = 0; t < 8; ++t) acc[t] = {};
  for (int k0 = 0; k0 < K; k0 += 32) {
    v16h af = frag_a(X, ldx, m0, k0, lane);
    v16h bf[8];
#pragma unroll
    for (int t = 0; t < 8; ++t) bf[t] = frag_b(W, ldw, t * 16, k0, lane);
#pragma unroll
    for (int t = 0; t < 8; ++t) acc[t] = wmma_f16(af, bf[t], acc[t]);
  }
  int mrow = m0 + ((lane >> 4) << 3);
  int nc = lane & 15;
#pragma unroll
  for (int t = 0; t < 8; ++t) {
    int n = t * 16 + nc;
    float bv = bias ? bias[n] : 0.f;
#pragma unroll
    for (int r = 0; r < 8; ++r) {
      float v = acc[t][r] + bv;
      if (act) v = lrelu(v);
      int m = mrow + r;
      if (outL) outL[m * LD + n] = (h16)v;
      if (outG) outG[(size_t)(grow + m) * 128 + n] = (h16)v;
    }
  }
}

// ---------------- kernel 1: BN partial stats ----------------
__global__ void __launch_bounds__(128) stats_partial_kernel(
    const float* __restrict__ obs, const float* __restrict__ acs,
    float* __restrict__ partial) {
  int a = blockIdx.y, chunk = blockIdx.x, f = threadIdx.x; // 128 threads
  size_t base = (size_t)a * NB + (size_t)chunk * 512;
  float s = 0.f, q = 0.f;
  for (int r = 0; r < 512; ++r) {
    size_t b = base + r;
    float v = (f < SD) ? obs[b * SD + f] : acs[b * AD + (f - SD)];
    s += v; q += v * v;
  }
  size_t o = (((size_t)a * 64 + chunk) * 128 + f) * 2;
  partial[o] = s; partial[o + 1] = q;
}

// ---------------- kernel 2: finalize mean / rstd ----------------
__global__ void __launch_bounds__(128) stats_final_kernel(
    const float* __restrict__ partial, float* __restrict__ stats) {
  int a = blockIdx.x, f = threadIdx.x;
  float s = 0.f, q = 0.f;
  for (int c = 0; c < 64; ++c) {
    size_t o = (((size_t)a * 64 + c) * 128 + f) * 2;
    s += partial[o]; q += partial[o + 1];
  }
  float mean = s / (float)NB;
  float var  = q / (float)NB - mean * mean;
  stats[(a * 128 + f) * 2]     = mean;
  stats[(a * 128 + f) * 2 + 1] = rsqrtf(var + 1e-5f);
}

// ---------------- kernel 3: encoders + k/sel/v projections ----------------
__global__ void __launch_bounds__(256) encode_kernel(
    const float* __restrict__ obs, const float* __restrict__ acs,
    const float* __restrict__ enc_W, const float* __restrict__ enc_b,
    const float* __restrict__ s_W,  const float* __restrict__ s_b,
    const float* __restrict__ k_W,  const float* __restrict__ sel_W,
    const float* __restrict__ v_W,  const float* __restrict__ v_b,
    const float* __restrict__ stats,
    h16* __restrict__ s_enc_g, h16* __restrict__ keys_g,
    h16* __restrict__ sels_g,  h16* __restrict__ vals_g) {
  __shared__ __align__(16) h16 Xsa[128 * LD];
  __shared__ __align__(16) h16 SAE[128 * LD];
  __shared__ __align__(16) h16 SE [128 * LD];
  __shared__ __align__(16) h16 Wt [128 * LD];
  __shared__ float sm[128], sr[128], bias[128];

  int a = blockIdx.y;
  int b0 = blockIdx.x * 128;
  int tid = threadIdx.x, lane = tid & 31, wv = tid >> 5;
  long grow = (long)a * NB + b0;

  // warm L2 with the per-head projection weights (global_prefetch_b8)
  for (int i = tid; i < 512; i += 256) {
    __builtin_prefetch(k_W + i * 32, 0, 1);
    __builtin_prefetch(sel_W + i * 32, 0, 1);
    __builtin_prefetch(v_W + i * 32, 0, 1);
  }

  for (int f = tid; f < 128; f += 256) {
    sm[f] = stats[(a * 128 + f) * 2];
    sr[f] = stats[(a * 128 + f) * 2 + 1];
  }
  __syncthreads();

  // normalized sa tile + enc_W^T + enc_b
  for (int idx = tid; idx < 128 * 128; idx += 256) {
    int r = idx >> 7, f = idx & 127;
    size_t b = (size_t)a * NB + b0 + r;
    float v = (f < SD) ? obs[b * SD + f] : acs[b * AD + (f - SD)];
    Xsa[r * LD + f] = (h16)((v - sm[f]) * sr[f]);
  }
  for (int idx = tid; idx < 128 * 128; idx += 256) {
    int k = idx >> 7, n = idx & 127;
    Wt[n * LD + k] = (h16)enc_W[idx];
  }
  for (int f = tid; f < 128; f += 256) bias[f] = enc_b[f];
  __syncthreads();

  // sa_enc = lrelu(bn(sa) @ enc_W + enc_b)
  do_gemm(Xsa, LD, Wt, LD, 128, bias, true, SAE, nullptr, 0, wv, lane);
  __syncthreads();

  // s_W^T (96x128 -> Wt[n][k], k<96) + s_b
  for (int idx = tid; idx < SD * 128; idx += 256) {
    int k = idx >> 7, n = idx & 127;
    Wt[n * LD + k] = (h16)s_W[idx];
  }
  for (int f = tid; f < 128; f += 256) bias[f] = s_b[f];
  __syncthreads();

  // s_enc = lrelu(bn(obs) @ s_W + s_b)  (bn(obs) == first 96 cols of Xsa)
  do_gemm(Xsa, LD, Wt, LD, SD, bias, true, SE, s_enc_g, grow, wv, lane);
  __syncthreads();

  // keys = sa_enc @ Kw  (Kw[f][h*32+d] = k_W[h][f][d])
  for (int idx = tid; idx < 16384; idx += 256) {
    int d = idx & 31, f = (idx >> 5) & 127, hh = idx >> 12;
    Wt[((hh << 5) | d) * LD + f] = (h16)k_W[idx];
  }
  __syncthreads();
  do_gemm(SAE, LD, Wt, LD, 128, nullptr, false, nullptr, keys_g, grow, wv, lane);
  __syncthreads();

  // sels = s_enc @ SelW
  for (int idx = tid; idx < 16384; idx += 256) {
    int d = idx & 31, f = (idx >> 5) & 127, hh = idx >> 12;
    Wt[((hh << 5) | d) * LD + f] = (h16)sel_W[idx];
  }
  __syncthreads();
  do_gemm(SE, LD, Wt, LD, 128, nullptr, false, nullptr, sels_g, grow, wv, lane);
  __syncthreads();

  // vals = lrelu(sa_enc @ Vw + v_b)
  for (int idx = tid; idx < 16384; idx += 256) {
    int d = idx & 31, f = (idx >> 5) & 127, hh = idx >> 12;
    Wt[((hh << 5) | d) * LD + f] = (h16)v_W[idx];
  }
  for (int f = tid; f < 128; f += 256) bias[f] = v_b[f];  // [4][32] flat == h*32+d
  __syncthreads();
  do_gemm(SAE, LD, Wt, LD, 128, bias, true, nullptr, vals_g, grow, wv, lane);
}

// ---------------- kernel 4: masked softmax attention over agents ----------------
__global__ void __launch_bounds__(256) attn_kernel(
    const h16* __restrict__ keys, const h16* __restrict__ sels,
    const h16* __restrict__ vals, h16* __restrict__ other) {
  int tid = threadIdx.x, lane = tid & 31;
  int b = blockIdx.x * 8 + (tid >> 5);   // one wave per batch element
  int h = lane >> 3, i = lane & 7;       // lane = (head, agent-i)
  const float scale = 0.1767766952966369f; // 1/sqrt(32)

  // selector slice for (i, b, h): 32 halves as 4x 16B vectors
  const v8h* sp = (const v8h*)(sels + ((size_t)i * NB + b) * 128 + h * 32);
  v8h sv[4];
#pragma unroll
  for (int c = 0; c < 4; ++c) sv[c] = sp[c];

  float lg[8];
#pragma unroll
  for (int j = 0; j < 8; ++j) {
    const v8h* kp = (const v8h*)(keys + ((size_t)j * NB + b) * 128 + h * 32);
    float acc = 0.f;
#pragma unroll
    for (int c = 0; c < 4; ++c) {
      v8h kv = kp[c];
#pragma unroll
      for (int d = 0; d < 8; ++d) acc += (float)sv[c][d] * (float)kv[d];
    }
    lg[j] = acc * scale;
  }
  lg[i] = -1e9f;
  float mx = lg[0];
#pragma unroll
  for (int j = 1; j < 8; ++j) mx = fmaxf(mx, lg[j]);
  float sum = 0.f;
#pragma unroll
  for (int j = 0; j < 8; ++j) { lg[j] = __expf(lg[j] - mx); sum += lg[j]; }
  float inv = 1.f / sum;

  float o[32];
#pragma unroll
  for (int d = 0; d < 32; ++d) o[d] = 0.f;
#pragma unroll
  for (int j = 0; j < 8; ++j) {
    const v8h* vp = (const v8h*)(vals + ((size_t)j * NB + b) * 128 + h * 32);
    float w = lg[j] * inv;
#pragma unroll
    for (int c = 0; c < 4; ++c) {
      v8h vv = vp[c];
#pragma unroll
      for (int d = 0; d < 8; ++d) o[c * 8 + d] += w * (float)vv[d];
    }
  }
  v8h* op = (v8h*)(other + ((size_t)i * NB + b) * 128 + h * 32);
#pragma unroll
  for (int c = 0; c < 4; ++c) {
    v8h ov;
#pragma unroll
    for (int d = 0; d < 8; ++d) ov[d] = (h16)o[c * 8 + d];
    op[c] = ov;
  }
}

// ---------------- kernel 5: critic MLP + argmax gather ----------------
__global__ void __launch_bounds__(256) critic_kernel(
    const h16* __restrict__ s_enc_g, const h16* __restrict__ other_g,
    const float* __restrict__ c_W1, const float* __restrict__ c_b1,
    const float* __restrict__ c_W2, const float* __restrict__ c_b2,
    const float* __restrict__ acs, float* __restrict__ out) {
  __shared__ __align__(16) h16 Xc [64 * CLD];
  __shared__ __align__(16) h16 W1t[128 * CLD];
  __shared__ __align__(16) h16 H1 [64 * LD];
  __shared__ __align__(16) h16 W2t[32 * LD];
  __shared__ float allq[64 * 36];
  __shared__ float b1s[128], b2s[32];

  int a = blockIdx.y;
  int b0 = blockIdx.x * 64;
  int tid = threadIdx.x, lane = tid & 31, wv = tid >> 5;

  for (int i = tid; i < 1024; i += 256) __builtin_prefetch(c_W1 + i * 32, 0, 1);

  // critic_in = [s_enc | other] : async global->LDS, 16B chunks.
  // Row = 256 halves = 32 chunks (16 from s_enc, 16 from other).
  for (int idx = tid; idx < 64 * 32; idx += 256) {
    int r = idx >> 5, c = idx & 31;
    size_t row = (size_t)a * NB + b0 + r;
    const h16* g = (c < 16) ? (s_enc_g + row * 128 + c * 8)
                            : (other_g + row * 128 + (c - 16) * 8);
    copy16_g2l(Xc + r * CLD + c * 8, g);
  }
  for (int idx = tid; idx < 256 * 128; idx += 256) {
    int k = idx >> 7, n = idx & 127;
    W1t[n * CLD + k] = (h16)c_W1[idx];
  }
  for (int f = tid; f < 128; f += 256) b1s[f] = c_b1[f];
  wait_async0();          // ASYNCcnt drained before the barrier
  __syncthreads();

  // h1 = lrelu(Xc @ c_W1 + b1): M=64, N=128, K=256
  {
    int m0 = (wv & 3) * 16;
    int nb = (wv >> 2) * 64;
    v8f acc[4];
#pragma unroll
    for (int t = 0; t < 4; ++t) acc[t] = {};
    for (int k0 = 0; k0 < 256; k0 += 32) {
      v16h af = frag_a(Xc, CLD, m0, k0, lane);
      v16h bf[4];
#pragma unroll
      for (int t = 0; t < 4; ++t) bf[t] = frag_b(W1t, CLD, nb + t * 16, k0, lane);
#pragma unroll
      for (int t = 0; t < 4; ++t) acc[t] = wmma_f16(af, bf[t], acc[t]);
    }
    int mrow = m0 + ((lane >> 4) << 3);
    int nc = lane & 15;
#pragma unroll
    for (int t = 0; t < 4; ++t) {
      int n = nb + t * 16 + nc;
#pragma unroll
      for (int r = 0; r < 8; ++r)
        H1[(mrow + r) * LD + n] = (h16)lrelu(acc[t][r] + b1s[n]);
    }
  }
  __syncthreads();

  for (int idx = tid; idx < 128 * 32; idx += 256) {
    int k = idx >> 5, n = idx & 31;
    W2t[n * LD + k] = (h16)c_W2[idx];
  }
  for (int f = tid; f < 32; f += 256) b2s[f] = c_b2[f];
  __syncthreads();

  // all_q = H1 @ c_W2 + b2: M=64, N=32, K=128 (one 16x16 tile per wave)
  {
    int m0 = (wv & 3) * 16;
    int n0 = (wv >> 2) * 16;
    v8f acc = {};
    for (int k0 = 0; k0 < 128; k0 += 32) {
      v16h af = frag_a(H1, LD, m0, k0, lane);
      v16h bf = frag_b(W2t, LD, n0, k0, lane);
      acc = wmma_f16(af, bf, acc);
    }
    int mrow = m0 + ((lane >> 4) << 3);
    int nc = lane & 15;
#pragma unroll
    for (int r = 0; r < 8; ++r)
      allq[(mrow + r) * 36 + n0 + nc] = acc[r] + b2s[n0 + nc];
  }
  __syncthreads();

  // q = all_q[argmax(acs)]
  if (tid < 64) {
    int r = tid;
    size_t row = (size_t)a * NB + b0 + r;
    const float* ap = acs + row * AD;
    float best = ap[0]; int bi = 0;
#pragma unroll
    for (int d = 1; d < AD; ++d) {
      float v = ap[d];
      if (v > best) { best = v; bi = d; }
    }
    out[row] = allq[r * 36 + bi];
  }
}

extern "C" void kernel_launch(void* const* d_in, const int* in_sizes, int n_in,
                              void* d_out, int out_size, void* d_ws, size_t ws_size,
                              hipStream_t stream) {
  const float* obs   = (const float*)d_in[0];
  const float* acs   = (const float*)d_in[1];
  const float* enc_W = (const float*)d_in[2];
  const float* enc_b = (const float*)d_in[3];
  const float* s_W   = (const float*)d_in[4];
  const float* s_b   = (const float*)d_in[5];
  const float* k_W   = (const float*)d_in[6];
  const float* sel_W = (const float*)d_in[7];
  const float* v_W   = (const float*)d_in[8];
  const float* v_b   = (const float*)d_in[9];
  const float* c_W1  = (const float*)d_in[10];
  const float* c_b1  = (const float*)d_in[11];
  const float* c_W2  = (const float*)d_in[12];
  const float* c_b2  = (const float*)d_in[13];
  float* out = (float*)d_out;

  char* ws = (char*)d_ws;
  const size_t T = (size_t)NA * NB * 128 * sizeof(h16); // 64 MB per f16 tensor
  float* partial = (float*)ws;                          // 512 KB
  float* stats   = (float*)(ws + 524288);               // 8 KB
  h16* s_enc_g = (h16*)(ws + 532480);
  h16* keys_g  = (h16*)(ws + 532480 + 1 * T);
  h16* sels_g  = (h16*)(ws + 532480 + 2 * T);
  h16* vals_g  = (h16*)(ws + 532480 + 3 * T);
  h16* other_g = (h16*)(ws + 532480 + 4 * T);

  stats_partial_kernel<<<dim3(64, NA), 128, 0, stream>>>(obs, acs, partial);
  stats_final_kernel<<<NA, 128, 0, stream>>>(partial, stats);
  encode_kernel<<<dim3(NB / 128, NA), 256, 0, stream>>>(
      obs, acs, enc_W, enc_b, s_W, s_b, k_W, sel_W, v_W, v_b, stats,
      s_enc_g, keys_g, sels_g, vals_g);
  attn_kernel<<<NB / 8, 256, 0, stream>>>(keys_g, sels_g, vals_g, other_g);
  critic_kernel<<<dim3(NB / 64, NA), 256, 0, stream>>>(
      s_enc_g, other_g, c_W1, c_b1, c_W2, c_b2, acs, out);
}
